// AttnLayerV3_17403207483594
// MI455X (gfx1250) — compile-verified, matrix operand order checked by reading, not run on hardware
//
#include <hip/hip_runtime.h>

typedef unsigned short ushort_t;
typedef __attribute__((ext_vector_type(8)))  __bf16 v8bf;
typedef __attribute__((ext_vector_type(16))) __bf16 v16bf;
typedef __attribute__((ext_vector_type(8)))  float  v8f;
typedef __attribute__((ext_vector_type(4)))  int    vi4_;

#define B_  2
#define T_  1024
#define D_  2048
#define H_  16
#define KV_ 4
#define DH_ 128
#define NQKV_ 6144            // 4096 + 1024 + 1024
#define LAMBDA_INIT_ 0.35550906759096926f
#define ONE_MINUS_LI_ 0.6444909324090307f
#define QSCALE_ 0.08838834764831845f   // DH^-0.5
#define LOG2_ROPE_OVER_HALF_ 0.20762050593046015f  // log2(10000)/64

#if __has_builtin(__builtin_amdgcn_global_load_async_to_lds_b128) && \
    __has_builtin(__builtin_amdgcn_s_wait_asynccnt)
#define HAS_ASYNC_ 1
#else
#define HAS_ASYNC_ 0
#endif

union FragU { v16bf v; v8bf h[2]; };

static __device__ __forceinline__ ushort_t f2bf(float f) {
    unsigned u = __float_as_uint(f);
    unsigned r = (u + 0x7FFFu + ((u >> 16) & 1u)) >> 16;
    return (ushort_t)r;
}

static __device__ __forceinline__ v16bf load_frag(const ushort_t* base, int stride, int row, int c0) {
    FragU f;
    f.h[0] = *(const v8bf*)(const void*)(base + row * stride + c0);
    f.h[1] = *(const v8bf*)(const void*)(base + row * stride + c0 + 16);
    return f.v;
}

// 16-byte global -> LDS copy: CDNA5 async DMA path when available, sync fallback otherwise.
static __device__ __forceinline__ void g2l_b128(ushort_t* l, const ushort_t* g) {
#if HAS_ASYNC_
    void* gv = (void*)g;   // drop const, then addrspace-cast to the builtin's v4i pointers
    void* lv = (void*)l;
    __builtin_amdgcn_global_load_async_to_lds_b128(
        (__attribute__((address_space(1))) vi4_*)gv,
        (__attribute__((address_space(3))) vi4_*)lv, 0, 0);
#else
    *(uint4*)(void*)l = *(const uint4*)(const void*)g;
#endif
}
static __device__ __forceinline__ void wait_async_prev() {  // retire previous 4-instr batch
#if HAS_ASYNC_
    __builtin_amdgcn_s_wait_asynccnt(4);
#endif
}
static __device__ __forceinline__ void wait_async_all() {
#if HAS_ASYNC_
    __builtin_amdgcn_s_wait_asynccnt(0);
#endif
}

// ---------------------------------------------------------------- cast fp32 -> bf16
__global__ void cast_kernel(const float* __restrict__ src, ushort_t* __restrict__ dst, int n4) {
    int i = blockIdx.x * blockDim.x + threadIdx.x;
    if (i >= n4) return;
    float4 f = ((const float4*)src)[i];
    ushort4 o;
    o.x = f2bf(f.x); o.y = f2bf(f.y); o.z = f2bf(f.z); o.w = f2bf(f.w);
    ((ushort4*)dst)[i] = o;
}

// ---------------------------------------------------------------- bf16 WMMA GEMM: C[M,N] = A[M,K] * W[N,K]^T
// 256 threads (8 waves), tile 128x128, k-step 32, double-buffered LDS with async prefetch.
// wave -> 64x32 (4x2 wmma tiles).
__launch_bounds__(256, 1)
__global__ void gemm_bf16(const ushort_t* __restrict__ A, const ushort_t* __restrict__ W,
                          float* __restrict__ C, int K, int ldc) {
    extern __shared__ ushort_t lds[];   // 2 x (128x48 A + 128x48 B) = 49152 bytes
    const int tid = threadIdx.x, lane = tid & 31, wave = tid >> 5;
    const int m0 = blockIdx.y * 128, n0 = blockIdx.x * 128;
    const int mw = (wave >> 2) * 64, nw = (wave & 3) * 32;
    const int lrow = lane & 15, c0h = (lane >> 4) * 8;

    v8f acc[4][2] = {};

    auto issue_tile = [&](int p, int kb) {
        ushort_t* as = lds + p * 12288;
        ushort_t* bs = as + 6144;
#pragma unroll
        for (int i = 0; i < 2; ++i) {
            int c = tid + i * 256;
            int r = c >> 2, col = (c & 3) << 3;
            g2l_b128(as + r * 48 + col, A + (size_t)(m0 + r) * K + kb + col);
            g2l_b128(bs + r * 48 + col, W + (size_t)(n0 + r) * K + kb + col);
        }
    };

    issue_tile(0, 0);
    int p = 0;
    for (int kb = 0; kb < K; kb += 32) {
        const bool more = (kb + 32) < K;
        if (more) {
            issue_tile(p ^ 1, kb + 32);   // prefetch next tile while current completes
            wait_async_prev();
        } else {
            wait_async_all();
        }
        __syncthreads();

        const ushort_t* as = lds + p * 12288;
        const ushort_t* bs = as + 6144;
        v16bf af[4], bfv[2];
#pragma unroll
        for (int mi = 0; mi < 4; ++mi) af[mi] = load_frag(as, 48, mw + mi * 16 + lrow, c0h);
#pragma unroll
        for (int ni = 0; ni < 2; ++ni) bfv[ni] = load_frag(bs, 48, nw + ni * 16 + lrow, c0h);
#pragma unroll
        for (int mi = 0; mi < 4; ++mi)
#pragma unroll
            for (int ni = 0; ni < 2; ++ni)
                acc[mi][ni] = __builtin_amdgcn_wmma_f32_16x16x32_bf16(
                    false, af[mi], false, bfv[ni], (short)0, acc[mi][ni], false, false);
        __syncthreads();
        p ^= 1;
    }
#pragma unroll
    for (int mi = 0; mi < 4; ++mi)
#pragma unroll
        for (int ni = 0; ni < 2; ++ni)
#pragma unroll
            for (int vg = 0; vg < 8; ++vg) {
                int r = m0 + mw + mi * 16 + vg + ((lane >> 4) << 3);
                int c = n0 + nw + ni * 16 + (lane & 15);
                C[(size_t)r * ldc + c] = acc[mi][ni][vg];
            }
}

// ---------------------------------------------------------------- RoPE + split to [b][head][comp][t][d] bf16
__global__ void rope_kernel(const float* __restrict__ qkv, const int* __restrict__ pos,
                            ushort_t* __restrict__ dst, int nh, int col0, float scale) {
    int idx = blockIdx.x * blockDim.x + threadIdx.x;
    int total = B_ * T_ * nh * 2 * (DH_ / 2);
    if (idx >= total) return;
    int i    = idx & 63;
    int comp = (idx >> 6) & 1;
    int head = (idx >> 7) % nh;
    int rem  = (idx >> 7) / nh;
    int t    = rem % T_;
    int b    = rem / T_;
    const float* src = qkv + (size_t)(b * T_ + t) * NQKV_ + col0 + head * 256 + comp * 128 + 2 * i;
    float x1 = src[0], x2 = src[1];
    float inv = exp2f(-(float)i * LOG2_ROPE_OVER_HALF_);
    float fr = (float)pos[t] * inv;
    float cs = cosf(fr), sn = sinf(fr);
    float o1 = (x1 * cs - x2 * sn) * scale;
    float o2 = (x1 * sn + x2 * cs) * scale;
    ushort_t* d = dst + ((size_t)((b * nh + head) * 2 + comp) * T_ + t) * DH_ + 2 * i;
    d[0] = f2bf(o1);
    d[1] = f2bf(o2);
}

// ---------------------------------------------------------------- V -> V^T bf16: vt[b][kv][c(256)][t]
__global__ void vt_kernel(const float* __restrict__ qkv, ushort_t* __restrict__ vt) {
    int idx = blockIdx.x * blockDim.x + threadIdx.x;
    int total = B_ * KV_ * 256 * T_;
    if (idx >= total) return;
    int t = idx & (T_ - 1);
    int c = (idx >> 10) & 255;
    int kv = (idx >> 18) & 3;
    int b = idx >> 20;
    float v = qkv[(size_t)(b * T_ + t) * NQKV_ + 5120 + kv * 256 + c];
    vt[idx] = f2bf(v);
}

// ---------------------------------------------------------------- lambda scalar
__global__ void lam_kernel(const float* lq1, const float* lk1, const float* lq2,
                           const float* lk2, float* out) {
    __shared__ float s1[128], s2[128];
    int t = threadIdx.x;
    s1[t] = lq1[t] * lk1[t];
    s2[t] = lq2[t] * lk2[t];
    __syncthreads();
    for (int o = 64; o > 0; o >>= 1) {
        if (t < o) { s1[t] += s1[t + o]; s2[t] += s2[t + o]; }
        __syncthreads();
    }
    if (t == 0) out[0] = __expf(s1[0]) - __expf(s2[0]) + LAMBDA_INIT_;
}

// ---------------------------------------------------------------- differential flash attention + groupnorm
// grid (T/64, H, B); 256 threads. waves 0-3: component 1, waves 4-7: component 2, same 64 rows.
#define KSTR_ 136            // padded LDS stride (elems), 272B = 17*16B
#define CST_  260
__launch_bounds__(256, 1)
__global__ void attn_kernel(const ushort_t* __restrict__ q_r, const ushort_t* __restrict__ k_r,
                            const ushort_t* __restrict__ vt, const float* __restrict__ lam_p,
                            const float* __restrict__ gnw, const float* __restrict__ gnb,
                            ushort_t* __restrict__ o_attn) {
    extern __shared__ ushort_t lds[];
    ushort_t* kds1 = lds;                 // 128 x 136
    ushort_t* kds2 = lds + 17408;         // 128 x 136
    ushort_t* vds  = lds + 34816;         // 256 x 136  (V^T tile: [col][key])
    ushort_t* pds  = lds + 69632;         // 8 waves x 16 x 136

    const int tid = threadIdx.x, lane = tid & 31, wave = tid >> 5;
    const int comp = wave >> 2, wstrip = wave & 3;
    const int qb = blockIdx.x, h = blockIdx.y, b = blockIdx.z;
    const int kv = h >> 2;
    const int lrow = lane & 15, lhalf = lane >> 4, c0h = (lane >> 4) * 8;

    const ushort_t* qbase = q_r + (size_t)((b * H_ + h) * 2 + comp) * T_ * DH_;
    const ushort_t* k1b   = k_r + (size_t)((b * KV_ + kv) * 2 + 0) * T_ * DH_;
    const ushort_t* k2b   = k_r + (size_t)((b * KV_ + kv) * 2 + 1) * T_ * DH_;
    const ushort_t* vtb   = vt  + (size_t)(b * KV_ + kv) * 256 * T_;

    v8f o_acc[16] = {};
    float m_run[8], l_run[8];
#pragma unroll
    for (int vg = 0; vg < 8; ++vg) { m_run[vg] = -1e30f; l_run[vg] = 0.f; }

    const int qrow0 = qb * 64 + wstrip * 16;
    const int kb_count = (qb + 2) >> 1;

    for (int kb = 0; kb < kb_count; ++kb) {
        __syncthreads();
        // cooperative async loads: K1, K2 (128x128), V^T (256x128) -> 32 async instrs / wave
#pragma unroll
        for (int i = 0; i < 8; ++i) {
            int c = tid + i * 256;
            int r = c >> 4, col = (c & 15) << 3;
            g2l_b128(kds1 + r * KSTR_ + col, k1b + (size_t)(kb * 128 + r) * DH_ + col);
            g2l_b128(kds2 + r * KSTR_ + col, k2b + (size_t)(kb * 128 + r) * DH_ + col);
        }
#pragma unroll
        for (int i = 0; i < 16; ++i) {
            int c = tid + i * 256;
            int r = c >> 4, col = (c & 15) << 3;
            g2l_b128(vds + r * KSTR_ + col, vtb + (size_t)r * T_ + kb * 128 + col);
        }
        wait_async_all();
        __syncthreads();

        const ushort_t* kds = comp ? kds2 : kds1;
        v8f sacc[8] = {};
#pragma unroll
        for (int kc = 0; kc < 4; ++kc) {
            FragU qf;
            const ushort_t* qp = qbase + (size_t)(qrow0 + lrow) * DH_ + kc * 32 + c0h;
            qf.h[0] = *(const v8bf*)(const void*)qp;
            qf.h[1] = *(const v8bf*)(const void*)(qp + 16);
#pragma unroll
            for (int nt = 0; nt < 8; ++nt) {
                v16bf kf = load_frag(kds, KSTR_, nt * 16 + lrow, kc * 32 + c0h);
                sacc[nt] = __builtin_amdgcn_wmma_f32_16x16x32_bf16(
                    false, qf.v, false, kf, (short)0, sacc[nt], false, false);
            }
        }
        // causal mask (exact; no-op for fully-below-diagonal blocks)
#pragma unroll
        for (int nt = 0; nt < 8; ++nt) {
            int key = kb * 128 + nt * 16 + lrow;
#pragma unroll
            for (int vg = 0; vg < 8; ++vg) {
                int qr = qrow0 + vg + (lhalf << 3);
                if (key > qr) sacc[nt][vg] = -1e30f;
            }
        }
        // online softmax per row (rows live on (vg, half); reduce over 16 lanes)
        float alpha[8];
#pragma unroll
        for (int vg = 0; vg < 8; ++vg) {
            float mloc = sacc[0][vg];
#pragma unroll
            for (int nt = 1; nt < 8; ++nt) mloc = fmaxf(mloc, sacc[nt][vg]);
            for (int off = 1; off < 16; off <<= 1) mloc = fmaxf(mloc, __shfl_xor(mloc, off, 32));
            float mnew = fmaxf(m_run[vg], mloc);
            float a = __expf(m_run[vg] - mnew);
            m_run[vg] = mnew;
            float ls = 0.f;
#pragma unroll
            for (int nt = 0; nt < 8; ++nt) {
                float p = __expf(sacc[nt][vg] - mnew);
                sacc[nt][vg] = p;
                ls += p;
            }
            for (int off = 1; off < 16; off <<= 1) ls += __shfl_xor(ls, off, 32);
            l_run[vg] = l_run[vg] * a + ls;
            alpha[vg] = a;
        }
#pragma unroll
        for (int n2 = 0; n2 < 16; ++n2)
#pragma unroll
            for (int vg = 0; vg < 8; ++vg) o_acc[n2][vg] *= alpha[vg];

        // write P (bf16) to per-wave LDS strip, re-read as A-fragments
        ushort_t* pw = pds + wave * 16 * KSTR_;
#pragma unroll
        for (int nt = 0; nt < 8; ++nt)
#pragma unroll
            for (int vg = 0; vg < 8; ++vg)
                pw[(vg + (lhalf << 3)) * KSTR_ + nt * 16 + lrow] = f2bf(sacc[nt][vg]);
        asm volatile("s_wait_dscnt 0" ::: "memory");

#pragma unroll
        for (int kc = 0; kc < 4; ++kc) {
            v16bf pf = load_frag(pw, KSTR_, lrow, kc * 32 + c0h);
#pragma unroll
            for (int n2 = 0; n2 < 16; ++n2) {
                v16bf vf = load_frag(vds, KSTR_, n2 * 16 + lrow, kc * 32 + c0h);
                o_acc[n2] = __builtin_amdgcn_wmma_f32_16x16x32_bf16(
                    false, pf, false, vf, (short)0, o_acc[n2], false, false);
            }
        }
    }

    // ---- combine O1/l1 - lam*O2/l2 in LDS, then GroupNorm + affine ----
    const float lam = lam_p[0];
    float* cmb = (float*)(void*)lds;  // 64 x CST_
    __syncthreads();
    if (comp == 0) {
#pragma unroll
        for (int n2 = 0; n2 < 16; ++n2)
#pragma unroll
            for (int vg = 0; vg < 8; ++vg) {
                int r = wstrip * 16 + vg + (lhalf << 3);
                cmb[r * CST_ + n2 * 16 + lrow] = o_acc[n2][vg] / l_run[vg];
            }
    }
    __syncthreads();
    if (comp == 1) {
#pragma unroll
        for (int n2 = 0; n2 < 16; ++n2)
#pragma unroll
            for (int vg = 0; vg < 8; ++vg) {
                int r = wstrip * 16 + vg + (lhalf << 3);
                int idx = r * CST_ + n2 * 16 + lrow;
                cmb[idx] = cmb[idx] - lam * (o_acc[n2][vg] / l_run[vg]);
            }
    }
    __syncthreads();

    int row = tid >> 2;        // 0..63
    int part = tid & 3;        // 64 cols each
    float s = 0.f, s2 = 0.f;
    for (int j = 0; j < 64; ++j) {
        float v = cmb[row * CST_ + part * 64 + j];
        s += v; s2 += v * v;
    }
    s  += __shfl_xor(s, 1, 32);  s  += __shfl_xor(s, 2, 32);
    s2 += __shfl_xor(s2, 1, 32); s2 += __shfl_xor(s2, 2, 32);
    float mu = s * (1.f / 256.f);
    float var = s2 * (1.f / 256.f) - mu * mu;
    float rinv = rsqrtf(var + 1e-5f);
    int trow = qb * 64 + row;
    ushort_t* op = o_attn + (size_t)(b * T_ + trow) * 4096 + h * 256;
    for (int j = 0; j < 64; ++j) {
        int c = part * 64 + j;
        float v = (cmb[row * CST_ + c] - mu) * rinv;
        v = v * gnw[h * 256 + c] + gnb[h * 256 + c];
        op[c] = f2bf(v * ONE_MINUS_LI_);
    }
}

// ---------------------------------------------------------------- host
extern "C" void kernel_launch(void* const* d_in, const int* in_sizes, int n_in,
                              void* d_out, int out_size, void* d_ws, size_t ws_size,
                              hipStream_t stream) {
    const float* x   = (const float*)d_in[0];
    const float* Wq  = (const float*)d_in[1];
    const float* Wk  = (const float*)d_in[2];
    const float* Wv  = (const float*)d_in[3];
    const float* Wo  = (const float*)d_in[4];
    const float* lq1 = (const float*)d_in[5];
    const float* lk1 = (const float*)d_in[6];
    const float* lq2 = (const float*)d_in[7];
    const float* lk2 = (const float*)d_in[8];
    const float* gnw = (const float*)d_in[9];
    const float* gnb = (const float*)d_in[10];
    const int*   pos = (const int*)d_in[11];
    float* out = (float*)d_out;

    size_t off = 0;
    auto alloc = [&](size_t bytes) -> void* {
        void* p = (char*)d_ws + off;
        off += (bytes + 255) & ~(size_t)255;
        return p;
    };
    const size_t M = (size_t)B_ * T_;                         // 2048
    ushort_t* xb   = (ushort_t*)alloc(M * D_ * 2);            // 8 MB
    ushort_t* wqkv = (ushort_t*)alloc((size_t)NQKV_ * D_ * 2);// 24 MB
    ushort_t* wob  = (ushort_t*)alloc((size_t)D_ * 4096 * 2); // 16 MB
    float*    qkv  = (float*)alloc(M * NQKV_ * 4);            // 48 MB
    ushort_t* q_r  = (ushort_t*)alloc((size_t)B_ * H_ * 2 * T_ * DH_ * 2);
    ushort_t* k_r  = (ushort_t*)alloc((size_t)B_ * KV_ * 2 * T_ * DH_ * 2);
    ushort_t* vtb  = (ushort_t*)alloc((size_t)B_ * KV_ * 256 * T_ * 2);
    float*    lam  = (float*)alloc(256);
    ushort_t* o_attn = wqkv;   // alias: wqkv is dead after the QKV GEMM

    auto grid1 = [](size_t n) { return dim3((unsigned)((n + 255) / 256)); };

    // casts (fused QKV weight packing)
    cast_kernel<<<grid1(M * D_ / 4), 256, 0, stream>>>(x, xb, (int)(M * D_ / 4));
    cast_kernel<<<grid1((size_t)4096 * D_ / 4), 256, 0, stream>>>(Wq, wqkv, 4096 * D_ / 4);
    cast_kernel<<<grid1((size_t)1024 * D_ / 4), 256, 0, stream>>>(Wk, wqkv + (size_t)4096 * D_, 1024 * D_ / 4);
    cast_kernel<<<grid1((size_t)1024 * D_ / 4), 256, 0, stream>>>(Wv, wqkv + (size_t)5120 * D_, 1024 * D_ / 4);
    cast_kernel<<<grid1((size_t)D_ * 4096 / 4), 256, 0, stream>>>(Wo, wob, D_ * 4096 / 4);

    // fused QKV projection: (2048 x 2048) @ (6144 x 2048)^T -> fp32, double-buffered async pipeline
    gemm_bf16<<<dim3(NQKV_ / 128, (unsigned)(M / 128)), 256, 49152, stream>>>(
        xb, wqkv, qkv, D_, NQKV_);

    // RoPE + split (scale folded into Q), V transpose, lambda scalar
    rope_kernel<<<grid1((size_t)B_ * T_ * H_ * 2 * 64), 256, 0, stream>>>(qkv, pos, q_r, H_, 0, QSCALE_);
    rope_kernel<<<grid1((size_t)B_ * T_ * KV_ * 2 * 64), 256, 0, stream>>>(qkv, pos, k_r, KV_, 4096, 1.0f);
    vt_kernel<<<grid1((size_t)B_ * KV_ * 256 * T_), 256, 0, stream>>>(qkv, vtb);
    lam_kernel<<<1, 128, 0, stream>>>(lq1, lk1, lq2, lk2, lam);

    // differential flash attention + groupnorm (170 KB dynamic LDS)
    attn_kernel<<<dim3(T_ / 64, H_, B_), 256, 174080, stream>>>(
        q_r, k_r, vtb, lam, gnw, gnb, o_attn);

    // output projection: (2048 x 4096) @ (2048 x 4096)^T -> d_out fp32
    gemm_bf16<<<dim3(D_ / 128, (unsigned)(M / 128)), 256, 49152, stream>>>(
        o_attn, wob, out, 4096, D_);

    (void)in_sizes; (void)n_in; (void)out_size; (void)ws_size;
}
